// ChainComplexMessagePassingLayer_18889266167942
// MI455X (gfx1250) — matrix operand order repaired and checked by reading, not compiled, not run on hardware
//
#include <hip/hip_runtime.h>
#include <hip/hip_bf16.h>
#include <math.h>

#define NQN 131072
#define NCN 65536
#define NE  1048576
#define DD  128
#define HID 256

typedef __attribute__((ext_vector_type(16))) __bf16        v16bf;
typedef __attribute__((ext_vector_type(8)))  float         v8f;
typedef __attribute__((ext_vector_type(8)))  unsigned int  v8u;

// ---------- helpers ----------

__device__ __forceinline__ unsigned short f2bf(float f) {
  union { float f; unsigned u; } v; v.f = f;
  unsigned u = v.u;
  u += 0x7FFFu + ((u >> 16) & 1u);   // round-to-nearest-even
  return (unsigned short)(u >> 16);
}

// A-matrix fragment (16x32 bf16, MxK) from a row-major LDS tile.
// Per ISA 7.12.2: lane L holds row M=L%16; VGPR v holds K-pair
// K0 = 16*(v/4) + 8*(L/16) + 2*(v%4)  (consecutive pair -> one b32 load).
__device__ __forceinline__ v16bf load_a_frag(const unsigned short* tile, int stride,
                                             int kbase, int lane) {
  const int m  = lane & 15;
  const int hi = (lane >> 4) & 1;
  const unsigned short* rowp = tile + m * stride + kbase + 8 * hi;
  v8u u;
#pragma unroll
  for (int v = 0; v < 8; ++v) {
    const int off = 16 * (v >> 2) + 2 * (v & 3);
    u[v] = *(const unsigned*)(rowp + off);
  }
  return __builtin_bit_cast(v16bf, u);
}

// B-matrix fragment (32x16 bf16, KxN) from column-major (transposed) bf16 weights
// in global ws. Lane L holds column N; K = kbase + 8*(L/16) + e (e<8),
// K = kbase + 16 + 8*(L/16) + (e-8) (e>=8) -> two b128 loads.
__device__ __forceinline__ v16bf load_b_frag(const unsigned short* wt, int kdim,
                                             int nbase, int kbase, int lane) {
  const int n  = nbase + (lane & 15);
  const int k0 = kbase + 8 * ((lane >> 4) & 1);
  const uint4* q = (const uint4*)(wt + n * kdim + k0);
  const uint4 a = q[0];
  const uint4 b = q[2];
  v8u u;
  u[0] = a.x; u[1] = a.y; u[2] = a.z; u[3] = a.w;
  u[4] = b.x; u[5] = b.y; u[6] = b.z; u[7] = b.w;
  return __builtin_bit_cast(v16bf, u);
}

__device__ __forceinline__ v8f wmma_bf16(v16bf a, v16bf b, v8f c) {
  return __builtin_amdgcn_wmma_f32_16x16x32_bf16(false, a, false, b, (short)0, c,
                                                 false, false);
}

__device__ __forceinline__ float gelu_exact(float x) {
  return 0.5f * x * (1.0f + erff(x * 0.70710678118654752f));
}

// ---------- kernels ----------

// f32 [rows x cols] -> bf16 transposed (column-major) [cols x rows]
__global__ __launch_bounds__(256) void wconv_kernel(const float* __restrict__ src,
                                                    unsigned short* __restrict__ dst,
                                                    int rows, int cols) {
  const int idx = blockIdx.x * 256 + threadIdx.x;
  if (idx >= rows * cols) return;
  const int r = idx / cols, c = idx % cols;
  dst[c * rows + r] = f2bf(src[r * cols + c]);
}

__global__ __launch_bounds__(256) void zero_kernel(float* __restrict__ p, int n) {
  const int i = blockIdx.x * 256 + threadIdx.x;
  if (i < n) p[i] = 0.0f;
}

// Fused: LN(H) @ Wres -> proj ; H @ Wrel -> rel. 16 rows per block, 8 waves.
__global__ __launch_bounds__(256) void lnproj_kernel(
    const float* __restrict__ H, const float* __restrict__ lng,
    const float* __restrict__ lnb,
    const unsigned short* __restrict__ WresT, const unsigned short* __restrict__ WrelT,
    float* __restrict__ proj, float* __restrict__ rel) {
  __shared__ unsigned short lnT[16 * DD];
  __shared__ unsigned short rawT[16 * DD];
  __shared__ float red[256], red2[256];
  __shared__ float mu[16], rs[16];

  const int tid = threadIdx.x;
  const int row = tid >> 4, seg = tid & 15;
  const size_t grow = (size_t)blockIdx.x * 16 + row;

  float x[8];
  float s = 0.f, s2 = 0.f;
  const float* hp = H + grow * DD + seg * 8;
#pragma unroll
  for (int i = 0; i < 8; ++i) { x[i] = hp[i]; s += x[i]; s2 += x[i] * x[i]; }
  red[tid] = s; red2[tid] = s2;
  __syncthreads();
  if (tid < 16) {
    float a = 0.f, b = 0.f;
#pragma unroll
    for (int i = 0; i < 16; ++i) { a += red[tid * 16 + i]; b += red2[tid * 16 + i]; }
    const float m = a * (1.0f / 128.0f);
    const float var = b * (1.0f / 128.0f) - m * m;
    mu[tid] = m;
    rs[tid] = rsqrtf(var + 1e-5f);
  }
  __syncthreads();
  {
    const float m = mu[row], r = rs[row];
#pragma unroll
    for (int i = 0; i < 8; ++i) {
      const int c = seg * 8 + i;
      lnT[row * DD + c]  = f2bf((x[i] - m) * r * lng[c] + lnb[c]);
      rawT[row * DD + c] = f2bf(x[i]);
    }
  }
  __syncthreads();

  const int lane = tid & 31, wave = tid >> 5;
  const int colbase = wave * 16;
  v8f accp = {};
  v8f accr = {};
#pragma unroll
  for (int ks = 0; ks < 4; ++ks) {
    v16bf a0 = load_a_frag(lnT, DD, ks * 32, lane);
    v16bf b0 = load_b_frag(WresT, DD, colbase, ks * 32, lane);
    accp = wmma_bf16(a0, b0, accp);
    v16bf a1 = load_a_frag(rawT, DD, ks * 32, lane);
    v16bf b1 = load_b_frag(WrelT, DD, colbase, ks * 32, lane);
    accr = wmma_bf16(a1, b1, accr);
  }
  const int hi = (lane >> 4) & 1;
  const int col = colbase + (lane & 15);
#pragma unroll
  for (int r = 0; r < 8; ++r) {
    const size_t o = ((size_t)blockIdx.x * 16 + r + 8 * hi) * DD + col;
    proj[o] = accp[r];
    rel[o]  = accr[r];
  }
}

// Edge scatter: acc[dst] += rel[src] * deg_src[src]^{-1/2}. 32 threads per edge.
__global__ __launch_bounds__(256) void scatter_kernel(
    const float* __restrict__ rel, const int* __restrict__ esrc,
    const int* __restrict__ edst, const float* __restrict__ degsrc,
    float* __restrict__ acc) {
  const long long t = (long long)blockIdx.x * 256 + threadIdx.x;
  const int e = (int)(t >> 5);
  if (e >= NE) return;
  const int c4 = (int)(t & 31) * 4;
  const int src = esrc[e], dst = edst[e];
  const float d = degsrc[src];
  const float s = d > 0.f ? rsqrtf(d) : 0.f;
  const float4 v = *(const float4*)(rel + (size_t)src * DD + c4);
  float* ap = acc + (size_t)dst * DD + c4;
  __hip_atomic_fetch_add(ap + 0, v.x * s, __ATOMIC_RELAXED, __HIP_MEMORY_SCOPE_AGENT);
  __hip_atomic_fetch_add(ap + 1, v.y * s, __ATOMIC_RELAXED, __HIP_MEMORY_SCOPE_AGENT);
  __hip_atomic_fetch_add(ap + 2, v.z * s, __ATOMIC_RELAXED, __HIP_MEMORY_SCOPE_AGENT);
  __hip_atomic_fetch_add(ap + 3, v.w * s, __ATOMIC_RELAXED, __HIP_MEMORY_SCOPE_AGENT);
}

// Fused MLP: h = GELU([proj | gate*deg_dst^{-1/2}*acc] @ W1 + b1);
//            out = proj + h @ W2 + b2.  16 rows per block, 8 waves.
__global__ __launch_bounds__(256) void mlp_kernel(
    const float* __restrict__ proj, const float* __restrict__ acc,
    const float* __restrict__ degdst, const float* __restrict__ gate,
    const unsigned short* __restrict__ W1T, const float* __restrict__ b1,
    const unsigned short* __restrict__ W2T, const float* __restrict__ b2,
    float* __restrict__ out) {
  __shared__ unsigned short aT[16 * HID];
  __shared__ unsigned short hT[16 * HID];

  const int tid = threadIdx.x;
  const int row = tid >> 4, seg = tid & 15;
  const size_t grow = (size_t)blockIdx.x * 16 + row;
  {
    const float dd = degdst[grow];
    const float sc = (dd > 0.f ? rsqrtf(dd) : 0.f) * gate[0];
    const int c0 = seg * 16;
#pragma unroll
    for (int i = 0; i < 16; ++i) {
      const int c = c0 + i;
      const float v = (c < DD) ? proj[grow * DD + c]
                               : acc[grow * DD + (c - DD)] * sc;
      aT[row * HID + c] = f2bf(v);
    }
  }
  __syncthreads();

  const int lane = tid & 31, wave = tid >> 5;
  const int hi = (lane >> 4) & 1;

  // GEMM1: 16x256 @ 256x256 ; each wave owns 2 of 16 column tiles.
#pragma unroll
  for (int half = 0; half < 2; ++half) {
    const int n = wave + 8 * half;
    v8f c = {};
#pragma unroll
    for (int ks = 0; ks < 8; ++ks) {
      v16bf a = load_a_frag(aT, HID, ks * 32, lane);
      v16bf b = load_b_frag(W1T, HID, n * 16, ks * 32, lane);
      c = wmma_bf16(a, b, c);
    }
    const int col = n * 16 + (lane & 15);
    const float bb = b1[col];
#pragma unroll
    for (int r = 0; r < 8; ++r) {
      hT[(r + 8 * hi) * HID + col] = f2bf(gelu_exact(c[r] + bb));
    }
  }
  __syncthreads();

  // GEMM2: 16x256 @ 256x128 ; each wave owns 1 of 8 column tiles.
  {
    v8f c = {};
#pragma unroll
    for (int ks = 0; ks < 8; ++ks) {
      v16bf a = load_a_frag(hT, HID, ks * 32, lane);
      v16bf b = load_b_frag(W2T, HID, wave * 16, ks * 32, lane);
      c = wmma_bf16(a, b, c);
    }
    const int col = wave * 16 + (lane & 15);
    const float bb = b2[col];
#pragma unroll
    for (int r = 0; r < 8; ++r) {
      const size_t o = ((size_t)blockIdx.x * 16 + r + 8 * hi) * DD + col;
      out[o] = proj[o] + c[r] + bb;
    }
  }
}

// ---------- launcher ----------

extern "C" void kernel_launch(void* const* d_in, const int* in_sizes, int n_in,
                              void* d_out, int out_size, void* d_ws, size_t ws_size,
                              hipStream_t stream) {
  (void)in_sizes; (void)n_in; (void)out_size; (void)ws_size;

  const float* H_Q        = (const float*)d_in[0];
  const float* H_C        = (const float*)d_in[1];
  const int*   eQC_src    = (const int*)d_in[2];
  const int*   eQC_dst    = (const int*)d_in[3];
  const int*   eCQ_src    = (const int*)d_in[4];
  const int*   eCQ_dst    = (const int*)d_in[5];
  const float* deg_QC_src = (const float*)d_in[6];
  const float* deg_QC_dst = (const float*)d_in[7];
  const float* deg_CQ_src = (const float*)d_in[8];
  const float* deg_CQ_dst = (const float*)d_in[9];
  const float* ln_g_Q     = (const float*)d_in[10];
  const float* ln_b_Q     = (const float*)d_in[11];
  const float* Wres_Q     = (const float*)d_in[12];
  const float* ln_g_C     = (const float*)d_in[13];
  const float* ln_b_C     = (const float*)d_in[14];
  const float* Wres_C     = (const float*)d_in[15];
  const float* Wrel_QC    = (const float*)d_in[16];
  const float* Wrel_CQ    = (const float*)d_in[17];
  const float* gate_QC    = (const float*)d_in[18];
  const float* gate_CQ    = (const float*)d_in[19];
  const float* W1_Q       = (const float*)d_in[20];
  const float* b1_Q       = (const float*)d_in[21];
  const float* W2_Q       = (const float*)d_in[22];
  const float* b2_Q       = (const float*)d_in[23];
  const float* W1_C       = (const float*)d_in[24];
  const float* b1_C       = (const float*)d_in[25];
  const float* W2_C       = (const float*)d_in[26];
  const float* b2_C       = (const float*)d_in[27];

  char* ws = (char*)d_ws;
  size_t off = 0;
  auto aus = [&](size_t e) { unsigned short* p = (unsigned short*)(ws + off); off += e * 2; return p; };
  auto af  = [&](size_t e) { float* p = (float*)(ws + off); off += e * 4; return p; };

  unsigned short* WresQt = aus((size_t)DD * DD);
  unsigned short* WresCt = aus((size_t)DD * DD);
  unsigned short* WrelQt = aus((size_t)DD * DD);   // Wrel_QC
  unsigned short* WrelCt = aus((size_t)DD * DD);   // Wrel_CQ
  unsigned short* W1Qt   = aus((size_t)HID * HID);
  unsigned short* W1Ct   = aus((size_t)HID * HID);
  unsigned short* W2Qt   = aus((size_t)HID * DD);
  unsigned short* W2Ct   = aus((size_t)HID * DD);
  float* projQ = af((size_t)NQN * DD);
  float* projC = af((size_t)NCN * DD);
  float* relQ  = af((size_t)NQN * DD);
  float* relC  = af((size_t)NCN * DD);
  float* accQ  = af((size_t)NQN * DD);   // msg to Q (from C over eCQ)
  float* accC  = af((size_t)NCN * DD);   // msg to C (from Q over eQC)

  float* outQ = (float*)d_out;
  float* outC = outQ + (size_t)NQN * DD;

  // 1) weight convert + transpose to bf16 column-major
  wconv_kernel<<<(DD * DD + 255) / 256, 256, 0, stream>>>(Wres_Q, WresQt, DD, DD);
  wconv_kernel<<<(DD * DD + 255) / 256, 256, 0, stream>>>(Wres_C, WresCt, DD, DD);
  wconv_kernel<<<(DD * DD + 255) / 256, 256, 0, stream>>>(Wrel_QC, WrelQt, DD, DD);
  wconv_kernel<<<(DD * DD + 255) / 256, 256, 0, stream>>>(Wrel_CQ, WrelCt, DD, DD);
  wconv_kernel<<<(HID * HID + 255) / 256, 256, 0, stream>>>(W1_Q, W1Qt, HID, HID);
  wconv_kernel<<<(HID * HID + 255) / 256, 256, 0, stream>>>(W1_C, W1Ct, HID, HID);
  wconv_kernel<<<(HID * DD + 255) / 256, 256, 0, stream>>>(W2_Q, W2Qt, HID, DD);
  wconv_kernel<<<(HID * DD + 255) / 256, 256, 0, stream>>>(W2_C, W2Ct, HID, DD);

  // 2) zero scatter accumulators
  zero_kernel<<<(NQN * DD + 255) / 256, 256, 0, stream>>>(accQ, NQN * DD);
  zero_kernel<<<(NCN * DD + 255) / 256, 256, 0, stream>>>(accC, NCN * DD);

  // 3) fused LN+proj / rel GEMMs (WMMA)
  lnproj_kernel<<<NQN / 16, 256, 0, stream>>>(H_Q, ln_g_Q, ln_b_Q, WresQt, WrelQt,
                                              projQ, relQ);
  lnproj_kernel<<<NCN / 16, 256, 0, stream>>>(H_C, ln_g_C, ln_b_C, WresCt, WrelCt,
                                              projC, relC);

  // 4) symmetric-normalized scatter: Q->C and C->Q
  scatter_kernel<<<(int)(((long long)NE * 32) / 256), 256, 0, stream>>>(
      relQ, eQC_src, eQC_dst, deg_QC_src, accC);
  scatter_kernel<<<(int)(((long long)NE * 32) / 256), 256, 0, stream>>>(
      relC, eCQ_src, eCQ_dst, deg_CQ_src, accQ);

  // 5) fused MLP (WMMA) with dst-degree scaling, bias, GELU, residual
  mlp_kernel<<<NQN / 16, 256, 0, stream>>>(projQ, accQ, deg_CQ_dst, gate_CQ,
                                           W1Qt, b1_Q, W2Qt, b2_Q, outQ);
  mlp_kernel<<<NCN / 16, 256, 0, stream>>>(projC, accC, deg_QC_dst, gate_QC,
                                           W1Ct, b1_C, W2Ct, b2_C, outC);
}
